// BertSelfAttention_27393301414070
// MI455X (gfx1250) — matrix operand order repaired and checked by reading, not compiled, not run on hardware
//
#include <hip/hip_runtime.h>

typedef __attribute__((ext_vector_type(16))) _Float16 v16h;
typedef __attribute__((ext_vector_type(8)))  _Float16 v8h;
typedef __attribute__((ext_vector_type(8)))  float    v8f;

static constexpr int Bc = 2, Sc = 2048, Hc = 1024, NHc = 16, DHc = 64;
#define NEG_INF_F (-3.40282347e38f)

// ---------------------------------------------------------------- fp32->fp16
__global__ void cvt_kernel(const float* __restrict__ src,
                           _Float16* __restrict__ dst, int n) {
  int i = blockIdx.x * blockDim.x + threadIdx.x;
  if (i < n) dst[i] = (_Float16)src[i];
}

__device__ __forceinline__ v16h cat8(v8h lo, v8h hi) {
  return __builtin_shufflevector(lo, hi, 0,1,2,3,4,5,6,7,8,9,10,11,12,13,14,15);
}

// ---------------------------------------------------------------- QKV GEMM
// Out = (X @ W^T + bias) * scale, written per-head.
// transpose_out==0: Out[bh, s, d]   (Q, K)
// transpose_out==1: Out[bh, d, s]   (V transposed for PV B-fragments)
__global__ __launch_bounds__(128) void qkv_gemm_kernel(
    const _Float16* __restrict__ X,   // [B*S, H] f16
    const _Float16* __restrict__ W,   // [H, H]   f16 (row = out feature)
    const float* __restrict__ bias,   // [H]
    _Float16* __restrict__ Out,
    int transpose_out, float scale)
{
  const int wave = threadIdx.x >> 5;
  const int lane = threadIdx.x & 31;
  const int grp  = lane >> 4;        // lane half (0/1)
  const int lcol = lane & 15;
  const int mbase = blockIdx.x * 16;
  const int ncol  = blockIdx.y * 64 + wave * 16 + lcol;

  const _Float16* arow = X + (size_t)(mbase + lcol) * Hc;  // A row = lane%16
  const _Float16* brow = W + (size_t)ncol * Hc;            // B col = W row

  v8f c = {};
  #pragma unroll 4
  for (int kb = 0; kb < Hc; kb += 32) {
    // A 16x32 f16 layout: lanes 0-15 K={0..7,16..23}, lanes 16-31 K={8..15,24..31}
    v8h a0 = *(const v8h*)(arow + kb + grp * 8);
    v8h a1 = *(const v8h*)(arow + kb + 16 + grp * 8);
    v16h a = cat8(a0, a1);
    // B 32x16 f16 layout: same K order as A (lanes 0-15 K={0..7,16..23}, lanes 16-31 K={8..15,24..31})
    v16h b = cat8(*(const v8h*)(brow + kb + grp * 8),
                  *(const v8h*)(brow + kb + 16 + grp * 8));
    c = __builtin_amdgcn_wmma_f32_16x16x32_f16(false, a, false, b,
                                               (short)0, c, false, false);
  }
  const float bs = bias[ncol];
  const int hh = ncol >> 6, d = ncol & 63;
  #pragma unroll
  for (int v = 0; v < 8; ++v) {      // C: row m = v + 8*grp, col = lane%16
    int m = mbase + v + grp * 8;
    int bidx = m >> 11;              // / S
    int s = m & (Sc - 1);
    float val = (c[v] + bs) * scale;
    size_t bh = (size_t)bidx * NHc + hh;
    if (transpose_out) Out[(bh * DHc + d) * Sc + s] = (_Float16)val;
    else               Out[(bh * Sc + s) * DHc + d] = (_Float16)val;
  }
}

// ---------------------------------------------------------------- attention
// One wave owns 16 query rows of one (b,h); streams keys in blocks of 32,
// keeping BOTH softmax branches (global additive mask / local 0-1 mask) online.
__global__ __launch_bounds__(128) void attn_kernel(
    const _Float16* __restrict__ Q,    // [B*NH, S, DH] (pre-scaled by 1/8)
    const _Float16* __restrict__ K,    // [B*NH, S, DH]
    const _Float16* __restrict__ Vt,   // [B*NH, DH, S]
    const float* __restrict__ amask,   // [B, S] additive
    const int* __restrict__ lmask,     // [B, S, S] 0/1
    const float* __restrict__ gate,    // [B*NH, S]
    float* __restrict__ out)           // [B, S, H] f32
{
  __shared__ _Float16 pbuf[4][2][16 * 32];   // per-wave P staging (C->A layout)
  const int wave = threadIdx.x >> 5;
  const int lane = threadIdx.x & 31;
  const int grp  = lane >> 4;
  const int lcol = lane & 15;
  const int bh = blockIdx.y;
  const int b = bh / NHc, hh = bh % NHc;
  const int qbase = blockIdx.x * 64 + wave * 16;

  // Q A-fragments (d = WMMA K dim), held for the whole kernel
  const _Float16* qrow = Q + ((size_t)bh * Sc + qbase + lcol) * DHc;
  v16h qa0 = cat8(*(const v8h*)(qrow + grp * 8),
                  *(const v8h*)(qrow + 16 + grp * 8));
  v16h qa1 = cat8(*(const v8h*)(qrow + 32 + grp * 8),
                  *(const v8h*)(qrow + 48 + grp * 8));

  const _Float16* Kb = K + (size_t)bh * Sc * DHc;
  const _Float16* Vb = Vt + (size_t)bh * DHc * Sc;
  const float* am = amask + (size_t)b * Sc;
  const int* lmb = lmask + (size_t)b * Sc * Sc;

  v8f accg[4] = {}, accl[4] = {};
  float mg[8], lg[8], ml[8], ll[8];
  #pragma unroll
  for (int v = 0; v < 8; ++v) { mg[v] = NEG_INF_F; ml[v] = NEG_INF_F; lg[v] = 0.f; ll[v] = 0.f; }

  _Float16* pbg = &pbuf[wave][0][0];
  _Float16* pbl = &pbuf[wave][1][0];

  // online-softmax update + P@V for one branch
  auto softmax_pv = [&](float (&x0)[8], float (&x1)[8], float (&mrun)[8],
                        float (&lrun)[8], v8f (&acc)[4], _Float16* pb,
                        const v16h (&vbf)[4]) {
    #pragma unroll
    for (int v = 0; v < 8; ++v) {
      float mx = fmaxf(x0[v], x1[v]);
      mx = fmaxf(mx, __shfl_xor(mx, 1, 32));
      mx = fmaxf(mx, __shfl_xor(mx, 2, 32));
      mx = fmaxf(mx, __shfl_xor(mx, 4, 32));
      mx = fmaxf(mx, __shfl_xor(mx, 8, 32));
      float mnew = fmaxf(mrun[v], mx);
      float fac = __expf(mrun[v] - mnew);
      mrun[v] = mnew;
      float p0 = __expf(x0[v] - mnew);
      float p1 = __expf(x1[v] - mnew);
      float rs = p0 + p1;
      rs += __shfl_xor(rs, 1, 32);
      rs += __shfl_xor(rs, 2, 32);
      rs += __shfl_xor(rs, 4, 32);
      rs += __shfl_xor(rs, 8, 32);
      lrun[v] = lrun[v] * fac + rs;
      #pragma unroll
      for (int nb = 0; nb < 4; ++nb) acc[nb][v] *= fac;
      // C layout (row = v+8*grp, col = lane%16) -> row-major 16x32 in LDS
      pb[(v + 8 * grp) * 32 + lcol]      = (_Float16)p0;
      pb[(v + 8 * grp) * 32 + 16 + lcol] = (_Float16)p1;
    }
    // read back in A layout (row = lane%16, K blocks of 8 per lane half)
    const _Float16* pr = pb + lcol * 32;
    v16h pa = cat8(*(const v8h*)(pr + grp * 8),
                   *(const v8h*)(pr + 16 + grp * 8));
    #pragma unroll
    for (int nb = 0; nb < 4; ++nb)
      acc[nb] = __builtin_amdgcn_wmma_f32_16x16x32_f16(false, pa, false, vbf[nb],
                                                       (short)0, acc[nb], false, false);
  };

  for (int jb = 0; jb < Sc; jb += 32) {
    // K^T B-fragments: col = key (lane%16), K dim = d in blocks of 8 per lane half (matches Q A)
    const _Float16* k0 = Kb + (size_t)(jb + lcol) * DHc + grp * 8;
    const _Float16* k1 = Kb + (size_t)(jb + 16 + lcol) * DHc + grp * 8;
    v16h kb00 = cat8(*(const v8h*)(k0), *(const v8h*)(k0 + 16));
    v16h kb01 = cat8(*(const v8h*)(k0 + 32), *(const v8h*)(k0 + 48));
    v16h kb10 = cat8(*(const v8h*)(k1), *(const v8h*)(k1 + 16));
    v16h kb11 = cat8(*(const v8h*)(k1 + 32), *(const v8h*)(k1 + 48));
    if (jb + 32 < Sc) {
      __builtin_prefetch(k0 + 32 * DHc, 0, 1);   // global_prefetch_b8
      __builtin_prefetch(k1 + 32 * DHc, 0, 1);
    }

    v8f s0 = {}, s1 = {};
    s0 = __builtin_amdgcn_wmma_f32_16x16x32_f16(false, qa0, false, kb00, (short)0, s0, false, false);
    s0 = __builtin_amdgcn_wmma_f32_16x16x32_f16(false, qa1, false, kb01, (short)0, s0, false, false);
    s1 = __builtin_amdgcn_wmma_f32_16x16x32_f16(false, qa0, false, kb10, (short)0, s1, false, false);
    s1 = __builtin_amdgcn_wmma_f32_16x16x32_f16(false, qa1, false, kb11, (short)0, s1, false, false);

    // V B-fragments for P@V: col = d, K dim = key in blocks of 8 per lane half (matches P A)
    v16h vbf[4];
    #pragma unroll
    for (int nb = 0; nb < 4; ++nb)
      vbf[nb] = cat8(*(const v8h*)(Vb + (size_t)(nb * 16 + lcol) * Sc + jb + grp * 8),
                     *(const v8h*)(Vb + (size_t)(nb * 16 + lcol) * Sc + jb + 16 + grp * 8));

    const float am0 = am[jb + lcol];
    const float am1 = am[jb + 16 + lcol];

    float x0[8], x1[8];
    // ---- global branch (additive mask per key column) ----
    #pragma unroll
    for (int v = 0; v < 8; ++v) { x0[v] = s0[v] + am0; x1[v] = s1[v] + am1; }
    softmax_pv(x0, x1, mg, lg, accg, pbg, vbf);

    // ---- local branch (0/1 mask -> -FLT_MAX additive, per element) ----
    #pragma unroll
    for (int v = 0; v < 8; ++v) {
      const int* lq = lmb + (size_t)(qbase + v + 8 * grp) * Sc + jb;
      float a0 = lq[lcol]      ? 0.f : NEG_INF_F;
      float a1 = lq[16 + lcol] ? 0.f : NEG_INF_F;
      x0[v] = s0[v] + a0;
      x1[v] = s1[v] + a1;
    }
    softmax_pv(x0, x1, ml, ll, accl, pbl, vbf);
  }

  // ---- finalize: gate blend, write [B,S,H] f32 ----
  const float* gp = gate + (size_t)bh * Sc + qbase;
  float g[8];
  #pragma unroll
  for (int v = 0; v < 8; ++v) g[v] = gp[v + 8 * grp];
  float* orow = out + ((size_t)b * Sc + qbase) * Hc + hh * DHc;
  #pragma unroll
  for (int nb = 0; nb < 4; ++nb) {
    #pragma unroll
    for (int v = 0; v < 8; ++v) {
      float cg = accg[nb][v] / lg[v];
      float cl = accl[nb][v] / ll[v];
      float o = g[v] * cl + (1.f - g[v]) * cg;
      orow[(size_t)(v + 8 * grp) * Hc + nb * 16 + lcol] = o;
    }
  }
}

// ---------------------------------------------------------------- launch
extern "C" void kernel_launch(void* const* d_in, const int* in_sizes, int n_in,
                              void* d_out, int out_size, void* d_ws, size_t ws_size,
                              hipStream_t stream) {
  const float* hidden = (const float*)d_in[0];
  const float* amask  = (const float*)d_in[1];
  const int*   lmask  = (const int*)d_in[2];
  const float* gate   = (const float*)d_in[3];
  const float* Wq = (const float*)d_in[4];
  const float* bq = (const float*)d_in[5];
  const float* Wk = (const float*)d_in[6];
  const float* bk = (const float*)d_in[7];
  const float* Wv = (const float*)d_in[8];
  const float* bv = (const float*)d_in[9];
  float* out = (float*)d_out;

  const size_t nX = (size_t)Bc * Sc * Hc;        // 4M elems
  const size_t nW = (size_t)Hc * Hc;             // 1M elems
  const size_t nQ = (size_t)Bc * NHc * Sc * DHc; // 4M elems

  _Float16* Xh  = (_Float16*)d_ws;
  _Float16* Wqh = Xh + nX;
  _Float16* Wkh = Wqh + nW;
  _Float16* Wvh = Wkh + nW;
  _Float16* Qh  = Wvh + nW;
  _Float16* Kh  = Qh + nQ;
  _Float16* Vth = Kh + nQ;   // 38 MB total workspace

  cvt_kernel<<<dim3((unsigned)((nX + 255) / 256)), dim3(256), 0, stream>>>(hidden, Xh, (int)nX);
  cvt_kernel<<<dim3((unsigned)((nW + 255) / 256)), dim3(256), 0, stream>>>(Wq, Wqh, (int)nW);
  cvt_kernel<<<dim3((unsigned)((nW + 255) / 256)), dim3(256), 0, stream>>>(Wk, Wkh, (int)nW);
  cvt_kernel<<<dim3((unsigned)((nW + 255) / 256)), dim3(256), 0, stream>>>(Wv, Wvh, (int)nW);

  dim3 gg(Bc * Sc / 16, Hc / 64);  // (256, 16), 4 waves/block
  qkv_gemm_kernel<<<gg, dim3(128), 0, stream>>>(Xh, Wqh, bq, Qh, 0, 0.125f); // fold 1/sqrt(64)
  qkv_gemm_kernel<<<gg, dim3(128), 0, stream>>>(Xh, Wkh, bk, Kh, 0, 1.0f);
  qkv_gemm_kernel<<<gg, dim3(128), 0, stream>>>(Xh, Wvh, bv, Vth, 1, 1.0f);  // V transposed

  attn_kernel<<<dim3(Sc / 64, Bc * NHc), dim3(128), 0, stream>>>(
      Qh, Kh, Vth, amask, lmask, gate, out);
}
